// HierarchicalPooling_43344809951970
// MI455X (gfx1250) — matrix operand order repaired and checked by reading, not compiled, not run on hardware
//
#include <hip/hip_runtime.h>
#include <hip/hip_bf16.h>
#include <math.h>

// ---------------------------------------------------------------------------
// Problem constants (from reference): H=256, B=4, N=5000, E=80000
// ---------------------------------------------------------------------------
#define HDIM   256
#define HHALF  128
#define NNODE  5000
#define BATCH  4
#define NEDGE  80000
#define K2     512          // 2*H (concat src|tgt)
#define NCHUNK 20           // node-pool parallel chunks
#define LDS_STRIDE 520      // halves per W1^T row (512 + 8 pad, stride = 260 dwords -> 4 mod 64 banks)

typedef __attribute__((ext_vector_type(16))) _Float16 v16h;
typedef __attribute__((ext_vector_type(8)))  _Float16 v8h;
typedef __attribute__((ext_vector_type(8)))  float    v8f;
typedef __attribute__((ext_vector_type(4)))  float    v4f;

// ---------------------------------------------------------------------------
// Kernel 1: node attention logits  na_logit[b,n] = tanh(x@nW1 + nb1)@nW2 + nb2
// one block (128 threads) per (b,n) row; x row staged in LDS
// ---------------------------------------------------------------------------
__global__ void __launch_bounds__(HHALF)
node_logits_kernel(const float* __restrict__ x,
                   const float* __restrict__ nW1,
                   const float* __restrict__ nb1,
                   const float* __restrict__ nW2,
                   const float* __restrict__ nb2,
                   float* __restrict__ nlogit) {
  __shared__ float xr[HDIM];
  __shared__ float red[HHALF];
  const int bn = blockIdx.x;                  // 0 .. B*N-1
  const int t  = threadIdx.x;                 // 0 .. 127
  const float* row = x + (size_t)bn * HDIM;
  xr[t]         = row[t];
  xr[t + HHALF] = row[t + HHALF];
  __syncthreads();

  float acc = nb1[t];
  #pragma unroll 8
  for (int i = 0; i < HDIM; ++i)
    acc = fmaf(xr[i], nW1[i * HHALF + t], acc);
  red[t] = tanhf(acc) * nW2[t];
  __syncthreads();
  #pragma unroll
  for (int s = 64; s > 0; s >>= 1) {
    if (t < s) red[t] += red[t + s];
    __syncthreads();
  }
  if (t == 0) nlogit[bn] = red[0] + nb2[0];
}

// ---------------------------------------------------------------------------
// Kernel 2: softmax over N per batch -> normalized weights nw[b,n]
// ---------------------------------------------------------------------------
__global__ void __launch_bounds__(1024)
node_softmax_kernel(const float* __restrict__ nlogit,
                    float* __restrict__ nw) {
  __shared__ float red[1024];
  const int b = blockIdx.x, t = threadIdx.x;
  const float* l = nlogit + (size_t)b * NNODE;

  float m = -3.4e38f;
  for (int n = t; n < NNODE; n += 1024) m = fmaxf(m, l[n]);
  red[t] = m; __syncthreads();
  for (int s = 512; s > 0; s >>= 1) { if (t < s) red[t] = fmaxf(red[t], red[t + s]); __syncthreads(); }
  m = red[0]; __syncthreads();

  float s = 0.f;
  for (int n = t; n < NNODE; n += 1024) s += __expf(l[n] - m);
  red[t] = s; __syncthreads();
  for (int ss = 512; ss > 0; ss >>= 1) { if (t < ss) red[t] += red[t + ss]; __syncthreads(); }
  const float inv = 1.f / red[0];
  __syncthreads();

  for (int n = t; n < NNODE; n += 1024)
    nw[(size_t)b * NNODE + n] = __expf(l[n] - m) * inv;
}

// ---------------------------------------------------------------------------
// Kernel 3: node pooling partial sums: partial[b,c,h] = sum_{n in chunk} nw*x
// ---------------------------------------------------------------------------
__global__ void __launch_bounds__(HDIM)
node_pool_partial(const float* __restrict__ x,
                  const float* __restrict__ nw,
                  float* __restrict__ partial) {
  const int b = blockIdx.x, c = blockIdx.y, h = threadIdx.x;
  const int n0 = c * (NNODE / NCHUNK), n1 = n0 + (NNODE / NCHUNK);
  float acc = 0.f;
  for (int n = n0; n < n1; ++n)
    acc = fmaf(nw[(size_t)b * NNODE + n], x[((size_t)b * NNODE + n) * HDIM + h], acc);
  partial[((size_t)b * NCHUNK + c) * HDIM + h] = acc;
}

__global__ void __launch_bounds__(HDIM)
node_pool_final(const float* __restrict__ partial, float* __restrict__ out) {
  const int b = blockIdx.x, h = threadIdx.x;
  float acc = 0.f;
  #pragma unroll
  for (int c = 0; c < NCHUNK; ++c)
    acc += partial[((size_t)b * NCHUNK + c) * HDIM + h];
  out[(size_t)b * (2 * HDIM) + h] = acc;        // node_pooled half of output
}

// ---------------------------------------------------------------------------
// Kernel 4 (hot): edge MLP via v_wmma_f32_16x16x32_f16.
// One wave handles 16 edges: A = [src|tgt] (16x512, f16 on the fly),
// B = sW1^T staged in LDS as f16, C accumulated f32 in TWO independent
// chains (even/odd K-chunks) to break the serial WMMA->WMMA C dependency,
// then fused tanh + sW2 dot. Writes per-edge logit and mean.
// ---------------------------------------------------------------------------
__global__ void __launch_bounds__(256)
edge_wmma_kernel(const float* __restrict__ x,
                 const int*  __restrict__ ei,
                 const float* __restrict__ sW1,
                 const float* __restrict__ sb1,
                 const float* __restrict__ sW2,
                 const float* __restrict__ sb2,
                 float* __restrict__ elogit,
                 float* __restrict__ emean) {
  extern __shared__ _Float16 ldsW[];   // W1^T: [256 cols][LDS_STRIDE halves of K]

  // cooperative transpose + f32->f16 convert of sW1 (512x256) into LDS
  for (int idx = threadIdx.x; idx < K2 * HDIM; idx += blockDim.x) {
    const int n = idx >> 9;            // output column 0..255
    const int k = idx & 511;           // input feature 0..511
    ldsW[n * LDS_STRIDE + k] = (_Float16)sW1[(size_t)k * HDIM + n];
  }
  __syncthreads();

  const int lane = threadIdx.x & 31;
  const int wave = threadIdx.x >> 5;
  const int m    = lane & 15;          // A row / B column within tile
  const int hi   = lane >> 4;          // half-wave K split
  const float b2 = sb2[0];

  const int nwaves = gridDim.x * (blockDim.x >> 5);
  const int gwave  = blockIdx.x * (blockDim.x >> 5) + wave;
  const int tilesPerBatch = NEDGE / 16;          // 5000, exact
  const int ntiles = BATCH * tilesPerBatch;

  for (int t = gwave; t < ntiles; t += nwaves) {
    const int b  = t / tilesPerBatch;
    const int e0 = (t % tilesPerBatch) * 16;
    const int e  = e0 + m;
    const int si = ei[e];
    const int ti = ei[NEDGE + e];
    const float* srow = x + ((size_t)b * NNODE + si) * HDIM;
    const float* trow = x + ((size_t)b * NNODE + ti) * HDIM;

    // ---- load A fragments (16 chunks of 16x32 f16), ISA 7.12.2 layout ----
    // lane (m,hi): halves K0+base+0..7 and K0+base+16..23, base = hi*8
    v16h afrag[16];
    float rsum = 0.f;                   // this lane's share of the row sum
    #pragma unroll
    for (int kc = 0; kc < 16; ++kc) {
      const int K0 = kc * 32;           // compile-time per unrolled iter
      const float* rowp = (K0 < HDIM) ? (srow + K0) : (trow + (K0 - HDIM));
      const int base = hi * 8;
      const v4f f0 = *(const v4f*)(rowp + base);
      const v4f f1 = *(const v4f*)(rowp + base + 4);
      const v4f f2 = *(const v4f*)(rowp + base + 16);
      const v4f f3 = *(const v4f*)(rowp + base + 20);
      v16h a;
      #pragma unroll
      for (int i = 0; i < 4; ++i) {
        a[i]      = (_Float16)f0[i];
        a[4 + i]  = (_Float16)f1[i];
        a[8 + i]  = (_Float16)f2[i];
        a[12 + i] = (_Float16)f3[i];
        rsum += f0[i] + f1[i] + f2[i] + f3[i];
      }
      afrag[kc] = a;
    }

    // ---- N-tiles: 16 cols each; C = A*B with dual accumulator chains ----
    float part[8] = {0.f, 0.f, 0.f, 0.f, 0.f, 0.f, 0.f, 0.f};
    #pragma unroll 1
    for (int nt = 0; nt < 16; ++nt) {
      const int ncol = nt * 16 + m;
      const _Float16* bp = &ldsW[ncol * LDS_STRIDE + hi * 8];
      v8f c0 = {};
      v8f c1 = {};
      #pragma unroll
      for (int kc = 0; kc < 16; kc += 2) {
        const v8h b0lo = *(const v8h*)(bp + kc * 32);
        const v8h b0hi = *(const v8h*)(bp + kc * 32 + 16);
        const v8h b1lo = *(const v8h*)(bp + kc * 32 + 32);
        const v8h b1hi = *(const v8h*)(bp + kc * 32 + 48);
        v16h bfrag0, bfrag1;
        #pragma unroll
        for (int i = 0; i < 8; ++i) {
          bfrag0[i] = b0lo[i]; bfrag0[8 + i] = b0hi[i];
          bfrag1[i] = b1lo[i]; bfrag1[8 + i] = b1hi[i];
        }
        c0 = __builtin_amdgcn_wmma_f32_16x16x32_f16(
                 false, afrag[kc],     false, bfrag0, (short)0, c0, false, false);
        c1 = __builtin_amdgcn_wmma_f32_16x16x32_f16(
                 false, afrag[kc + 1], false, bfrag1, (short)0, c1, false, false);
      }
      const float b1v = sb1[ncol];
      const float w2v = sW2[ncol];
      // C layout: VGPR r -> (M = r + hi*8, N = m)
      #pragma unroll
      for (int r = 0; r < 8; ++r)
        part[r] += tanhf((c0[r] + c1[r]) + b1v) * w2v;
    }

    // butterfly-reduce each part[r] across the 16 lanes of this half-wave
    #pragma unroll
    for (int r = 0; r < 8; ++r) {
      float v = part[r];
      v += __shfl_xor(v, 1, 32);
      v += __shfl_xor(v, 2, 32);
      v += __shfl_xor(v, 4, 32);
      v += __shfl_xor(v, 8, 32);
      part[r] = v;
    }
    #pragma unroll
    for (int r = 0; r < 8; ++r)
      if (m == r)
        elogit[(size_t)b * NEDGE + e0 + hi * 8 + r] = part[r] + b2;

    // edge_mean = (sum(src) + sum(tgt)) / 512 ; lanes m and m+16 hold halves
    const float full = rsum + __shfl_xor(rsum, 16, 32);
    if (hi == 0)
      emean[(size_t)b * NEDGE + e0 + m] = full * (1.f / 512.f);
  }
}

// ---------------------------------------------------------------------------
// Kernel 5: edge softmax over E + weighted mean -> broadcast scalar to output
// ---------------------------------------------------------------------------
__global__ void __launch_bounds__(1024)
edge_reduce_kernel(const float* __restrict__ elogit,
                   const float* __restrict__ emean,
                   float* __restrict__ out) {
  __shared__ float red[1024];
  __shared__ float red2[1024];
  const int b = blockIdx.x, t = threadIdx.x;
  const float* l  = elogit + (size_t)b * NEDGE;
  const float* mm = emean  + (size_t)b * NEDGE;

  float mx = -3.4e38f;
  for (int e = t; e < NEDGE; e += 1024) mx = fmaxf(mx, l[e]);
  red[t] = mx; __syncthreads();
  for (int s = 512; s > 0; s >>= 1) { if (t < s) red[t] = fmaxf(red[t], red[t + s]); __syncthreads(); }
  mx = red[0]; __syncthreads();

  float s1 = 0.f, s2 = 0.f;
  for (int e = t; e < NEDGE; e += 1024) {
    const float w = __expf(l[e] - mx);
    s1 += w; s2 = fmaf(w, mm[e], s2);
  }
  red[t] = s1; red2[t] = s2; __syncthreads();
  for (int s = 512; s > 0; s >>= 1) {
    if (t < s) { red[t] += red[t + s]; red2[t] += red2[t + s]; }
    __syncthreads();
  }
  const float sp = red2[0] / red[0];
  if (t < HDIM)
    out[(size_t)b * (2 * HDIM) + HDIM + t] = sp;   // structure_pooled half
}

// ---------------------------------------------------------------------------
extern "C" void kernel_launch(void* const* d_in, const int* in_sizes, int n_in,
                              void* d_out, int out_size, void* d_ws, size_t ws_size,
                              hipStream_t stream) {
  const float* x   = (const float*)d_in[0];
  const int*   ei  = (const int*)  d_in[1];
  const float* nW1 = (const float*)d_in[2];
  const float* nb1 = (const float*)d_in[3];
  const float* nW2 = (const float*)d_in[4];
  const float* nb2 = (const float*)d_in[5];
  const float* sW1 = (const float*)d_in[6];
  const float* sb1 = (const float*)d_in[7];
  const float* sW2 = (const float*)d_in[8];
  const float* sb2 = (const float*)d_in[9];
  float* out = (float*)d_out;

  float* ws     = (float*)d_ws;                 // ~2.8 MB total
  float* nlogit = ws;                           // B*N
  float* nw     = nlogit + BATCH * NNODE;       // B*N
  float* elogit = nw + BATCH * NNODE;           // B*E
  float* emean  = elogit + (size_t)BATCH * NEDGE;   // B*E
  float* npart  = emean  + (size_t)BATCH * NEDGE;   // B*NCHUNK*H

  node_logits_kernel<<<BATCH * NNODE, HHALF, 0, stream>>>(x, nW1, nb1, nW2, nb2, nlogit);
  node_softmax_kernel<<<BATCH, 1024, 0, stream>>>(nlogit, nw);
  node_pool_partial<<<dim3(BATCH, NCHUNK), HDIM, 0, stream>>>(x, nw, npart);
  node_pool_final<<<BATCH, HDIM, 0, stream>>>(npart, out);

  const size_t ldsBytes = (size_t)HDIM * LDS_STRIDE * sizeof(_Float16); // 266,240 B
  edge_wmma_kernel<<<320, 256, ldsBytes, stream>>>(x, ei, sW1, sb1, sW2, sb2, elogit, emean);
  edge_reduce_kernel<<<BATCH, 1024, 0, stream>>>(elogit, emean, out);
}
